// RelativePosAttentionLayer_52355651338436
// MI455X (gfx1250) — matrix operand order repaired and checked by reading, not compiled, not run on hardware
//
#include <hip/hip_runtime.h>

#define SLEN 2048
#define NH   16
#define DH   64
#define HIDD 1024

typedef __bf16 v16bf __attribute__((ext_vector_type(16)));
typedef __bf16 v8bf  __attribute__((ext_vector_type(8)));
typedef float  v8f   __attribute__((ext_vector_type(8)));
typedef int    v4i_vs __attribute__((vector_size(4 * sizeof(int))));

union BF16x16 { v16bf v; v8bf h8[2]; };

static __device__ __forceinline__ v8f wmma_bf16(v16bf a, v16bf b, v8f c) {
  // D = A(16x32 bf16) * B(32x16 bf16) + C(16x16 f32)
  return __builtin_amdgcn_wmma_f32_16x16x32_bf16(false, a, false, b, (short)0, c, false, false);
}

// Async global->LDS copy (GLOBAL_LOAD_ASYNC_TO_LDS_B128, ASYNCcnt) when the
// toolchain declares the builtin; otherwise a synchronous load+ds_store pair.
#if defined(__HIP_DEVICE_COMPILE__) && __has_builtin(__builtin_amdgcn_global_load_async_to_lds_b128)
#define HAVE_ASYNC_LDS 1
#else
#define HAVE_ASYNC_LDS 0
#endif

static __device__ __forceinline__ void cp16_to_lds(__bf16* dst, const __bf16* src) {
#if HAVE_ASYNC_LDS
  __builtin_amdgcn_global_load_async_to_lds_b128(
      (__attribute__((address_space(1))) v4i_vs*)src,
      (__attribute__((address_space(3))) v4i_vs*)dst, 0, 0);
#else
  *(v8bf*)dst = *(const v8bf*)src;
#endif
}

static __device__ __forceinline__ void async_wait0() {
#if HAVE_ASYNC_LDS
  asm volatile("s_wait_asynccnt 0" ::: "memory");
#endif
}

// ---------------------------------------------------------------------------
// Phase 1: fused projection GEMM  out = X @ W.T + b  (f32 in -> bf16 out)
// Q,K stored head-major [B,H,S,DH]; V stored transposed [B,H,DH,S].
// One wave computes a 16x64 tile of the 4096x1024 output.
// ---------------------------------------------------------------------------
__global__ __launch_bounds__(256) void proj_kernel(
    const float* __restrict__ query, const float* __restrict__ key_, const float* __restrict__ value,
    const float* __restrict__ Wq, const float* __restrict__ bq,
    const float* __restrict__ Wk, const float* __restrict__ bk,
    const float* __restrict__ Wv, const float* __restrict__ bv,
    __bf16* __restrict__ Qh, __bf16* __restrict__ Kh, __bf16* __restrict__ VhT)
{
  const int pz = blockIdx.z;
  const float* X  = (pz == 0) ? query : (pz == 1) ? key_ : value;
  const float* W  = (pz == 0) ? Wq    : (pz == 1) ? Wk   : Wv;
  const float* Bb = (pz == 0) ? bq    : (pz == 1) ? bk   : bv;

  const int wave = threadIdx.x >> 5;
  const int lane = threadIdx.x & 31;
  const int n16  = lane & 15;
  const int half = lane >> 4;
  const int m0 = (blockIdx.x * 8 + wave) * 16;   // row tile in [0,4096)
  const int n0 = blockIdx.y * 64;                // col tile in [0,1024)

  v8f acc[4] = {};

  for (int kk = 0; kk < HIDD; kk += 32) {
    // A 16x32: lane row m=n16; half0 holds K kk+{0..15}, half1 kk+{16..31}
    BF16x16 a;
    const float* ap = X + (size_t)(m0 + n16) * HIDD + kk + 16 * half;
    #pragma unroll
    for (int i = 0; i < 8; i++) { a.v[i] = (__bf16)ap[i]; a.v[8 + i] = (__bf16)ap[8 + i]; }
    if (kk + 32 < HIDD) __builtin_prefetch(ap + 32);
    #pragma unroll
    for (int t = 0; t < 4; t++) {
      // B 32x16: element (k, n) = W[n, k]; lane col n=n16, half selects K 0..15 / 16..31
      const float* bp = W + (size_t)(n0 + t * 16 + n16) * HIDD + kk + 16 * half;
      BF16x16 bb;
      #pragma unroll
      for (int i = 0; i < 16; i++) bb.v[i] = (__bf16)bp[i];
      acc[t] = wmma_bf16(a.v, bb.v, acc[t]);
    }
  }

  #pragma unroll
  for (int t = 0; t < 4; t++) {
    const int gn   = n0 + t * 16 + n16;
    const float bi = Bb[gn];
    const int hh = gn >> 6, d = gn & 63;
    #pragma unroll
    for (int r = 0; r < 8; r++) {
      const int gm = m0 + r + 8 * half;            // C layout: m = r + 8*half
      const int bb_ = gm >> 11, s = gm & (SLEN - 1);
      const float val = acc[t][r] + bi;
      if (pz == 0)      Qh [((size_t)(bb_ * NH + hh) * SLEN + s) * DH + d]  = (__bf16)val;
      else if (pz == 1) Kh [((size_t)(bb_ * NH + hh) * SLEN + s) * DH + d]  = (__bf16)val;
      else              VhT[((size_t)(bb_ * NH + hh) * DH + d) * SLEN + s]  = (__bf16)val;
    }
  }
}

// ---------------------------------------------------------------------------
// Phase 2: fused rel-pos attention (no softmax in reference -> fully linear,
// stream over k tiles). One wave owns a 16-row q tile; the 8 waves of a block
// share one (b,h), so K/V tiles are staged cooperatively in LDS with double
// buffering (async global->LDS when available).
// ---------------------------------------------------------------------------
__global__ __launch_bounds__(256) void attn_kernel(
    const __bf16* __restrict__ Qh, const __bf16* __restrict__ Kh, const __bf16* __restrict__ VhT,
    const float* __restrict__ rk, const float* __restrict__ rv, float* __restrict__ out)
{
  __shared__ float  rvs[5 * DH];          // rel_v table, block-wide
  __shared__ __bf16 Kst[2][32 * DH];      // staged K tile [k_local][d], double-buffered
  __shared__ __bf16 Vst[2][DH * 32];      // staged V^T tile [d][k_local], double-buffered
  __shared__ __bf16 s_lds[8][16 * 32];    // per-wave attn tile (16x32)
  __shared__ float  qds[8][16 * 8];       // per-wave qdot[m][j] (j<5 used)
  __shared__ float  snear[8][64];         // per-wave near-diagonal buckets [m][3]

  const int tid = threadIdx.x;
  for (int i = tid; i < 5 * DH; i += 256) rvs[i] = rv[i];

  const int wave = tid >> 5, lane = tid & 31;
  const int n16 = lane & 15, half = lane >> 4;
  snear[wave][lane] = 0.f; snear[wave][lane + 32] = 0.f;

  const int b = blockIdx.z, h = blockIdx.y;
  const int q0 = (blockIdx.x * 8 + wave) * 16;
  const __bf16* Q = Qh  + (size_t)(b * NH + h) * SLEN * DH;
  const __bf16* K = Kh  + (size_t)(b * NH + h) * SLEN * DH;
  const __bf16* V = VhT + (size_t)(b * NH + h) * DH * SLEN;

  // Q tile as two A matrices (d = 0..31, 32..63)
  BF16x16 aq[2];
  #pragma unroll
  for (int c = 0; c < 2; c++) {
    const v8bf* p = (const v8bf*)(Q + (size_t)(q0 + n16) * DH + 32 * c + 16 * half);
    aq[c].h8[0] = p[0];
    aq[c].h8[1] = p[1];   // +8 bf16
  }

  // qdot[m][j] = qh[m,:] . rel_k[j,:]  via WMMA (B cols >=5 are zero)
  v8f qa = {};
  #pragma unroll
  for (int c = 0; c < 2; c++) {
    BF16x16 brk;
    #pragma unroll
    for (int i = 0; i < 16; i++) brk.v[i] = (__bf16)0.0f;
    if (n16 < 5) {
      const float* rp = rk + n16 * DH + 32 * c + 16 * half;
      #pragma unroll
      for (int i = 0; i < 16; i++) brk.v[i] = (__bf16)rp[i];
    }
    qa = wmma_bf16(aq[c].v, brk.v, qa);
  }
  if (n16 < 5) {
    #pragma unroll
    for (int r = 0; r < 8; r++) qds[wave][(r + 8 * half) * 8 + n16] = qa[r];
  }
  asm volatile("s_wait_dscnt 0" ::: "memory");

  float qd0[8], qd4[8];
  #pragma unroll
  for (int r = 0; r < 8; r++) { int m = r + 8 * half; qd0[r] = qds[wave][m * 8]; qd4[r] = qds[wave][m * 8 + 4]; }

  float slo[8], shi[8];
  #pragma unroll
  for (int r = 0; r < 8; r++) { slo[r] = 0.f; shi[r] = 0.f; }
  v8f O[4] = {};
  __bf16* sl = s_lds[wave];

  // Stage first K/V tile cooperatively: 256 threads x 16B each per tensor.
  const int dd = tid >> 2, pp = tid & 3;
  cp16_to_lds(&Kst[0][tid * 8], K + tid * 8);
  cp16_to_lds(&Vst[0][dd * 32 + pp * 8], V + (size_t)dd * SLEN + pp * 8);
  async_wait0();
  __syncthreads();

  int buf = 0;
  #pragma unroll 1
  for (int kt0 = 0; kt0 < SLEN; kt0 += 32) {
    const int nbuf = buf ^ 1;
    if (kt0 + 32 < SLEN) {   // prefetch next tile into the other buffer
      cp16_to_lds(&Kst[nbuf][tid * 8], K + (size_t)(kt0 + 32) * DH + tid * 8);
      cp16_to_lds(&Vst[nbuf][dd * 32 + pp * 8], V + (size_t)dd * SLEN + (kt0 + 32) + pp * 8);
    }
    const __bf16* Kb = Kst[buf];
    const __bf16* Vb = Vst[buf];

    #pragma unroll
    for (int c2 = 0; c2 < 2; c2++) {
      const int kc = kt0 + 16 * c2;
      v8f sa = {};
      #pragma unroll
      for (int c = 0; c < 2; c++) {
        // B = K^T chunk: element (dk, kcol) = Kst[kcol_local][32c+dk]
        BF16x16 bk;
        const v8bf* kp = (const v8bf*)(Kb + (size_t)(c2 * 16 + n16) * DH + 32 * c + 16 * half);
        bk.h8[0] = kp[0]; bk.h8[1] = kp[1];
        sa = wmma_bf16(aq[c].v, bk.v, sa);
      }
      const int kcol = kc + n16;
      if (kc <= q0 - 32) {            // all elements have k <= q-2 -> j=0
        #pragma unroll
        for (int r = 0; r < 8; r++) {
          float a = (sa[r] + qd0[r]) * 0.125f;
          slo[r] += a;
          sl[(r + 8 * half) * 32 + c2 * 16 + n16] = (__bf16)a;
        }
      } else if (kc >= q0 + 32) {     // all elements have k >= q+2 -> j=4
        #pragma unroll
        for (int r = 0; r < 8; r++) {
          float a = (sa[r] + qd4[r]) * 0.125f;
          shi[r] += a;
          sl[(r + 8 * half) * 32 + c2 * 16 + n16] = (__bf16)a;
        }
      } else {                        // near-diagonal tile: element-wise clip
        #pragma unroll
        for (int r = 0; r < 8; r++) {
          const int m = r + 8 * half, q = q0 + m;
          int dl = kcol - q;
          int dc = dl < -2 ? -2 : (dl > 2 ? 2 : dl);
          float a = (sa[r] + qds[wave][m * 8 + dc + 2]) * 0.125f;
          if (dl <= -2)      slo[r] += a;
          else if (dl >= 2)  shi[r] += a;
          else               atomicAdd(&snear[wave][m * 4 + (dl + 1)], a);
          sl[m * 32 + c2 * 16 + n16] = (__bf16)a;
        }
      }
    }
    asm volatile("s_wait_dscnt 0" ::: "memory");
    // Reload S tile in A layout from LDS
    BF16x16 as_;
    {
      const v8bf* sp = (const v8bf*)(sl + n16 * 32 + 16 * half);
      as_.h8[0] = sp[0];
      as_.h8[1] = sp[1];
    }
    #pragma unroll
    for (int t = 0; t < 4; t++) {
      // B = V chunk: element (kr, n) = Vst[t*16+n][kr]
      BF16x16 bv16;
      const v8bf* vp = (const v8bf*)(Vb + (size_t)(t * 16 + n16) * 32 + 16 * half);
      bv16.h8[0] = vp[0]; bv16.h8[1] = vp[1];
      O[t] = wmma_bf16(as_.v, bv16.v, O[t]);
    }

    async_wait0();       // next tile's async fill done (per issuing wave)
    __syncthreads();     // all waves done with buf, nbuf visible block-wide
    buf = nbuf;
  }

  __syncthreads();

  // Row-sum reduction across the 16 lanes of each half (rows shared per half)
  #pragma unroll
  for (int r = 0; r < 8; r++) {
    #pragma unroll
    for (int msk = 1; msk < 16; msk <<= 1) {
      slo[r] += __shfl_xor(slo[r], msk, 32);
      shi[r] += __shfl_xor(shi[r], msk, 32);
    }
  }
  float s1v[8], s2v[8], s3v[8];
  #pragma unroll
  for (int r = 0; r < 8; r++) {
    int m = r + 8 * half;
    s1v[r] = snear[wave][m * 4 + 0];
    s2v[r] = snear[wave][m * 4 + 1];
    s3v[r] = snear[wave][m * 4 + 2];
  }

  #pragma unroll
  for (int t = 0; t < 4; t++) {
    const int d = t * 16 + n16;
    const float r0 = rvs[d], r1 = rvs[DH + d], r2 = rvs[2 * DH + d], r3 = rvs[3 * DH + d], r4 = rvs[4 * DH + d];
    #pragma unroll
    for (int r = 0; r < 8; r++) {
      const int m = r + 8 * half;
      const float w2 = slo[r] * r0 + s1v[r] * r1 + s2v[r] * r2 + s3v[r] * r3 + shi[r] * r4;
      out[(((size_t)b * SLEN + q0 + m) * NH + h) * DH + d] = O[t][r] + w2;
    }
  }
}

// ---------------------------------------------------------------------------
extern "C" void kernel_launch(void* const* d_in, const int* in_sizes, int n_in,
                              void* d_out, int out_size, void* d_ws, size_t ws_size,
                              hipStream_t stream) {
  (void)in_sizes; (void)n_in; (void)out_size; (void)ws_size;
  const float* query = (const float*)d_in[0];
  const float* key_  = (const float*)d_in[1];
  const float* value = (const float*)d_in[2];
  const float* Wq = (const float*)d_in[3];
  const float* bq = (const float*)d_in[4];
  const float* Wk = (const float*)d_in[5];
  const float* bk = (const float*)d_in[6];
  const float* Wv = (const float*)d_in[7];
  const float* bv = (const float*)d_in[8];
  const float* rk = (const float*)d_in[9];
  const float* rv = (const float*)d_in[10];

  __bf16* ws = (__bf16*)d_ws;
  const size_t QKN = (size_t)2 * NH * SLEN * DH;   // 4,194,304 elems per tensor
  __bf16* Qh  = ws;
  __bf16* Kh  = ws + QKN;
  __bf16* VhT = ws + 2 * QKN;                      // 24 MB of ws total (bf16)

  proj_kernel<<<dim3(32, 16, 3), 256, 0, stream>>>(query, key_, value,
                                                   Wq, bq, Wk, bk, Wv, bv,
                                                   Qh, Kh, VhT);
  attn_kernel<<<dim3(16, 16, 2), 256, 0, stream>>>(Qh, Kh, VhT, rk, rv, (float*)d_out);
}
